// DeepSeekMoE_45784351375732
// MI455X (gfx1250) — matrix operand order repaired and checked by reading, not compiled, not run on hardware
//
#include <hip/hip_runtime.h>
#include <hip/hip_bf16.h>
#include <math.h>

typedef __attribute__((ext_vector_type(16))) __bf16 v16bf;
typedef __attribute__((ext_vector_type(8)))  __bf16 v8bf;
typedef __attribute__((ext_vector_type(4)))  __bf16 v4bf;
typedef __attribute__((ext_vector_type(8)))  float  v8f;
typedef __attribute__((ext_vector_type(4)))  float  v4f;

// Problem constants (from reference: B=4,S=2048,H=1024,E=8,I=2048,K=2)
constexpr int   T_  = 8192;
constexpr int   H_  = 1024;
constexpr int   E_  = 8;
constexpr int   I_  = 2048;
constexpr int   H1ROWS = T_ * 2 + E_ * 32;   // 32-padded per-expert capacity bound

// ---------------------------------------------------------------- utilities
__global__ void k_zero4(float* __restrict__ p) {
    int i = (blockIdx.x * blockDim.x + threadIdx.x) * 4;
    v4f z = {};
    *(v4f*)(p + i) = z;
}

__global__ void k_cvt_x(const float* __restrict__ x, __bf16* __restrict__ xb) {
    int i = (blockIdx.x * blockDim.x + threadIdx.x) * 4;
    v4f v = *(const v4f*)(x + i);
    v4bf o;
    #pragma unroll
    for (int j = 0; j < 4; ++j) o[j] = (__bf16)v[j];
    *(v4bf*)(xb + i) = o;
}

// in: [E][R][C] f32  ->  out: [E][C][R] bf16   (32x32 LDS tiles, block 32x8)
__global__ void k_transpose_cvt(const float* __restrict__ in, __bf16* __restrict__ out,
                                int R, int C) {
    __shared__ float tile[32][33];
    const int e  = blockIdx.z;
    const int c0 = blockIdx.x * 32;
    const int r0 = blockIdx.y * 32;
    const float* src = in  + (size_t)e * R * C;
    __bf16*      dst = out + (size_t)e * R * C;
    #pragma unroll
    for (int rr = 0; rr < 4; ++rr) {
        int r = r0 + threadIdx.y + rr * 8;
        tile[threadIdx.y + rr * 8][threadIdx.x] = src[(size_t)r * C + c0 + threadIdx.x];
    }
    __syncthreads();
    #pragma unroll
    for (int rr = 0; rr < 4; ++rr) {
        int c = c0 + threadIdx.y + rr * 8;
        dst[(size_t)c * R + r0 + threadIdx.x] = (__bf16)tile[threadIdx.x][threadIdx.y + rr * 8];
    }
}

// ---------------------------------------------------------------- gating
// one wave32 per token: logits = x[t,:] @ Wg, softmax, top-2, renormalize
__global__ void k_gate(const float* __restrict__ x, const float* __restrict__ Wg,
                       float* __restrict__ w) {
    const int wave = threadIdx.x >> 5;
    const int lane = threadIdx.x & 31;
    const int t = blockIdx.x * 4 + wave;
    const float* xr = x + (size_t)t * H_;
    float acc[E_] = {};
    for (int j = lane; j < H_; j += 32) {
        float xv = xr[j];
        const float* wr = Wg + (size_t)j * E_;
        #pragma unroll
        for (int e = 0; e < E_; ++e) acc[e] += xv * wr[e];
    }
    #pragma unroll
    for (int e = 0; e < E_; ++e) {
        #pragma unroll
        for (int off = 16; off > 0; off >>= 1) acc[e] += __shfl_xor(acc[e], off, 32);
    }
    if (lane == 0) {
        float m = acc[0];
        #pragma unroll
        for (int e = 1; e < E_; ++e) m = fmaxf(m, acc[e]);
        float p[E_];
        #pragma unroll
        for (int e = 0; e < E_; ++e) p[e] = __expf(acc[e] - m);
        int i1 = 0;
        #pragma unroll
        for (int e = 1; e < E_; ++e) if (p[e] > p[i1]) i1 = e;          // ties -> lower idx
        int i2 = (i1 == 0) ? 1 : 0;
        #pragma unroll
        for (int e = 0; e < E_; ++e) if (e != i1 && p[e] > p[i2]) i2 = e;
        float denom = p[i1] + p[i2];
        float* wr = w + (size_t)t * E_;
        #pragma unroll
        for (int e = 0; e < E_; ++e) wr[e] = 0.f;
        wr[i1] = p[i1] / denom;
        wr[i2] = p[i2] / denom;
    }
}

// ---------------------------------------------------------------- routing
// one wave per expert; ballot/prefix append preserves ascending token order
__global__ void k_route(const float* __restrict__ w, int* __restrict__ lists,
                        int* __restrict__ counts) {
    const int e = blockIdx.x;
    const int lane = threadIdx.x;
    int cnt = 0;
    int* list = lists + (size_t)e * T_;
    for (int base = 0; base < T_; base += 32) {
        int t = base + lane;
        bool pred = w[(size_t)t * E_ + e] > 0.f;
        unsigned mask = (unsigned)__ballot(pred);
        if (pred) {
            int prefix = __popc(mask & ((1u << lane) - 1u));
            list[cnt + prefix] = t;
        }
        cnt += __popc(mask);
    }
    if (lane == 0) counts[e] = cnt;
}

__global__ void k_offsets(const int* __restrict__ counts, int* __restrict__ offs) {
    if (threadIdx.x == 0 && blockIdx.x == 0) {
        int off = 0;
        for (int e = 0; e < E_; ++e) { offs[e] = off; off += (counts[e] + 31) & ~31; }
    }
}

// ---------------------------------------------------------------- WMMA helpers
__device__ __forceinline__ v16bf pack_frag(v8bf lo, v8bf hi) {
    v16bf f;
    #pragma unroll
    for (int j = 0; j < 8; ++j) { f[j] = lo[j]; f[j + 8] = hi[j]; }
    return f;
}

// ---------------------------------------------------------------- GEMM1 + GELU
// h1[row, :I] = gelu(x_gather[row,:H] @ W1t[e]^T + b1[e]) ; W1t: [E][I][H] bf16
// 2M x 2N register tile per wave: 4 independent WMMA chains, 8 b128 loads / 4 wmma
__global__ __launch_bounds__(256)
void k_expert_gemm1(const __bf16* __restrict__ xb, const __bf16* __restrict__ W1t,
                    const float* __restrict__ b1, const int* __restrict__ lists,
                    const int* __restrict__ counts, const int* __restrict__ offs,
                    __bf16* __restrict__ h1) {
    const int e = blockIdx.z;
    const int count = counts[e];
    const int mtile = blockIdx.y;                          // 32 rows per block
    if (mtile * 32 >= count) return;                       // block-uniform exit
    const int wave = threadIdx.x >> 5, lane = threadIdx.x & 31;
    const int half = lane >> 4;                            // K-half select
    const int n0 = (blockIdx.x * 8 + wave) * 32 + (lane & 15);

    int rl0 = mtile * 32 + (lane & 15);
    int rl1 = rl0 + 16;
    if (rl0 >= count) rl0 = count - 1;                     // pad rows: duplicate
    if (rl1 >= count) rl1 = count - 1;
    const int t0 = lists[(size_t)e * T_ + rl0];
    const int t1 = lists[(size_t)e * T_ + rl1];

    const __bf16* arow0 = xb + (size_t)t0 * H_ + half * 8;
    const __bf16* arow1 = xb + (size_t)t1 * H_ + half * 8;
    const __bf16* brow0 = W1t + ((size_t)e * I_ + n0) * H_ + half * 8;
    const __bf16* brow1 = brow0 + (size_t)16 * H_;

    v8f acc00 = {}, acc01 = {}, acc10 = {}, acc11 = {};
    for (int k0 = 0; k0 < H_; k0 += 32) {
        v16bf fa0 = pack_frag(*(const v8bf*)(arow0 + k0), *(const v8bf*)(arow0 + k0 + 16));
        v16bf fa1 = pack_frag(*(const v8bf*)(arow1 + k0), *(const v8bf*)(arow1 + k0 + 16));
        v16bf fb0 = pack_frag(*(const v8bf*)(brow0 + k0), *(const v8bf*)(brow0 + k0 + 16));
        v16bf fb1 = pack_frag(*(const v8bf*)(brow1 + k0), *(const v8bf*)(brow1 + k0 + 16));
        acc00 = __builtin_amdgcn_wmma_f32_16x16x32_bf16(false, fa0, false, fb0, (short)0, acc00, false, false);
        acc01 = __builtin_amdgcn_wmma_f32_16x16x32_bf16(false, fa0, false, fb1, (short)0, acc01, false, false);
        acc10 = __builtin_amdgcn_wmma_f32_16x16x32_bf16(false, fa1, false, fb0, (short)0, acc10, false, false);
        acc11 = __builtin_amdgcn_wmma_f32_16x16x32_bf16(false, fa1, false, fb1, (short)0, acc11, false, false);
    }

    const int hbase = offs[e] + mtile * 32;
    #pragma unroll
    for (int mt = 0; mt < 2; ++mt) {
        #pragma unroll
        for (int tt = 0; tt < 2; ++tt) {
            const int n = n0 + tt * 16;
            const float bias = b1[(size_t)e * I_ + n];
            v8f a = mt ? (tt ? acc11 : acc10) : (tt ? acc01 : acc00);
            #pragma unroll
            for (int r = 0; r < 8; ++r) {
                int rr = mt * 16 + r + half * 8;
                float v = a[r] + bias;
                v = 0.5f * v * (1.0f + erff(v * 0.70710678118654752f));  // exact erf GELU
                h1[(size_t)(hbase + rr) * I_ + n] = (__bf16)v;
            }
        }
    }
}

// ---------------------------------------------------------------- GEMM2 + combine
// out[t,:] += w[t,e] * (h1[row,:I] @ W2t[e]^T + b2[e]) ; W2t: [E][H][I] bf16
__global__ __launch_bounds__(256)
void k_expert_gemm2(const __bf16* __restrict__ h1, const __bf16* __restrict__ W2t,
                    const float* __restrict__ b2, const float* __restrict__ w,
                    const int* __restrict__ lists, const int* __restrict__ counts,
                    const int* __restrict__ offs, float* __restrict__ out) {
    const int e = blockIdx.z;
    const int count = counts[e];
    const int mtile = blockIdx.y;                          // 32 rows per block
    if (mtile * 32 >= count) return;
    const int wave = threadIdx.x >> 5, lane = threadIdx.x & 31;
    const int half = lane >> 4;
    const int n0 = (blockIdx.x * 8 + wave) * 32 + (lane & 15);

    const int arowi = offs[e] + mtile * 32 + (lane & 15);  // padded rows always valid
    const __bf16* arow0 = h1 + (size_t)arowi * I_ + half * 8;
    const __bf16* arow1 = arow0 + (size_t)16 * I_;
    const __bf16* brow0 = W2t + ((size_t)e * H_ + n0) * I_ + half * 8;
    const __bf16* brow1 = brow0 + (size_t)16 * I_;

    v8f acc00 = {}, acc01 = {}, acc10 = {}, acc11 = {};
    for (int k0 = 0; k0 < I_; k0 += 32) {
        v16bf fa0 = pack_frag(*(const v8bf*)(arow0 + k0), *(const v8bf*)(arow0 + k0 + 16));
        v16bf fa1 = pack_frag(*(const v8bf*)(arow1 + k0), *(const v8bf*)(arow1 + k0 + 16));
        v16bf fb0 = pack_frag(*(const v8bf*)(brow0 + k0), *(const v8bf*)(brow0 + k0 + 16));
        v16bf fb1 = pack_frag(*(const v8bf*)(brow1 + k0), *(const v8bf*)(brow1 + k0 + 16));
        acc00 = __builtin_amdgcn_wmma_f32_16x16x32_bf16(false, fa0, false, fb0, (short)0, acc00, false, false);
        acc01 = __builtin_amdgcn_wmma_f32_16x16x32_bf16(false, fa0, false, fb1, (short)0, acc01, false, false);
        acc10 = __builtin_amdgcn_wmma_f32_16x16x32_bf16(false, fa1, false, fb0, (short)0, acc10, false, false);
        acc11 = __builtin_amdgcn_wmma_f32_16x16x32_bf16(false, fa1, false, fb1, (short)0, acc11, false, false);
    }

    #pragma unroll
    for (int mt = 0; mt < 2; ++mt) {
        #pragma unroll
        for (int tt = 0; tt < 2; ++tt) {
            const int n = n0 + tt * 16;
            const float bias = b2[(size_t)e * H_ + n];
            v8f a = mt ? (tt ? acc11 : acc10) : (tt ? acc01 : acc00);
            #pragma unroll
            for (int r = 0; r < 8; ++r) {
                int row_local = mtile * 32 + mt * 16 + r + half * 8;
                if (row_local < count) {
                    int tkn = lists[(size_t)e * T_ + row_local];
                    float scale = w[(size_t)tkn * E_ + e];
                    unsafeAtomicAdd(&out[(size_t)tkn * H_ + n], (a[r] + bias) * scale);
                }
            }
        }
    }
}

// ---------------------------------------------------------------- launch
extern "C" void kernel_launch(void* const* d_in, const int* in_sizes, int n_in,
                              void* d_out, int out_size, void* d_ws, size_t ws_size,
                              hipStream_t stream) {
    const float* x  = (const float*)d_in[0];   // [B,S,H]
    const float* Wg = (const float*)d_in[1];   // [H,E]
    const float* W1 = (const float*)d_in[2];   // [E,H,I]
    const float* b1 = (const float*)d_in[3];   // [E,I]
    const float* W2 = (const float*)d_in[4];   // [E,I,H]
    const float* b2 = (const float*)d_in[5];   // [E,H]
    float* out = (float*)d_out;                // [B,S,H]

    char* ws = (char*)d_ws;
    size_t off = 0;
    __bf16* xb   = (__bf16*)(ws + off); off += (size_t)T_ * H_ * 2;          // 16 MB
    __bf16* w1t  = (__bf16*)(ws + off); off += (size_t)E_ * I_ * H_ * 2;     // 32 MB
    __bf16* w2t  = (__bf16*)(ws + off); off += (size_t)E_ * H_ * I_ * 2;     // 32 MB
    __bf16* h1   = (__bf16*)(ws + off); off += (size_t)H1ROWS * I_ * 2;      // 65 MB
    float*  wcmb = (float*) (ws + off); off += (size_t)T_ * E_ * 4;
    int*    lists= (int*)   (ws + off); off += (size_t)E_ * T_ * 4;
    int*    cnts = (int*)   (ws + off); off += 128;
    int*    offs = (int*)   (ws + off); off += 128;

    // 1. zero output (harness poisons it)
    k_zero4<<<(T_ * H_ / 4) / 256, 256, 0, stream>>>(out);
    // 2. precision conversion / weight transposition
    k_cvt_x<<<(T_ * H_ / 4) / 256, 256, 0, stream>>>(x, xb);
    k_transpose_cvt<<<dim3(I_ / 32, H_ / 32, E_), dim3(32, 8), 0, stream>>>(W1, w1t, H_, I_);
    k_transpose_cvt<<<dim3(H_ / 32, I_ / 32, E_), dim3(32, 8), 0, stream>>>(W2, w2t, I_, H_);
    // 3. gating + deterministic routing
    k_gate<<<T_ / 4, 128, 0, stream>>>(x, Wg, wcmb);
    k_route<<<E_, 32, 0, stream>>>(wcmb, lists, cnts);
    k_offsets<<<1, 32, 0, stream>>>(cnts, offs);
    // 4. grouped GEMM1 (+GELU) and GEMM2 (+combine), 32-row M tiles
    k_expert_gemm1<<<dim3(I_ / 256, T_ / 32, E_), 256, 0, stream>>>(
        xb, w1t, b1, lists, cnts, offs, h1);
    k_expert_gemm2<<<dim3(H_ / 256, T_ / 32, E_), 256, 0, stream>>>(
        h1, w2t, b2, wcmb, lists, cnts, offs, out);
}
